// HybridTransformerLNNv5Model_52922587022085
// MI455X (gfx1250) — compile-verified, matrix operand order checked by reading, not run on hardware
//
#include <hip/hip_runtime.h>
#include <hip/hip_bf16.h>
#include <math.h>

typedef _Float16 h16;
typedef _Float16 v8h  __attribute__((ext_vector_type(8)));
typedef _Float16 v16h __attribute__((ext_vector_type(16)));
typedef float    v8f  __attribute__((ext_vector_type(8)));

#define LN_EPS 1e-5f

__device__ __forceinline__ float gelu_exact(float x) {
  return 0.5f * x * (1.0f + erff(x * 0.7071067811865475f));
}

__device__ __forceinline__ v16h combine16(v8h lo, v8h hi) {
  v16h r;
#pragma unroll
  for (int i = 0; i < 8; ++i) { r[i] = lo[i]; r[i + 8] = hi[i]; }
  return r;
}

__device__ __forceinline__ v8h zero8() {
  v8h z;
#pragma unroll
  for (int i = 0; i < 8; ++i) z[i] = (h16)0.0f;
  return z;
}

// ---------------------------------------------------------------- converts
__global__ void k_f32_to_f16(const float* __restrict__ in, h16* __restrict__ out, int n) {
  int i = blockIdx.x * blockDim.x + threadIdx.x;
  if (i < n) out[i] = (h16)in[i];
}

// in[R][C] (f32) -> out[C][R] (f16)
__global__ void k_trans_f32_to_f16(const float* __restrict__ in, h16* __restrict__ out, int R, int C) {
  int i = blockIdx.x * blockDim.x + threadIdx.x;
  if (i < R * C) {
    int r = i / C, c = i % C;
    out[(size_t)c * R + r] = (h16)in[i];
  }
}

// conv weight W[oc][ic][3] (f32) -> Wt[t][ic][oc] (f16)
__global__ void k_convprep(const float* __restrict__ W, h16* __restrict__ Wt, int OC, int IC) {
  int i = blockIdx.x * blockDim.x + threadIdx.x;
  int total = OC * IC * 3;
  if (i < total) {
    int oc = i / (IC * 3);
    int rem = i % (IC * 3);
    int ic = rem / 3;
    int t  = rem % 3;
    Wt[(size_t)t * IC * OC + (size_t)ic * OC + oc] = (h16)W[i];
  }
}

// fold BN(eval): sa = gamma*rsqrt(var+eps); sb = (b-mean)*sa + beta
__global__ void k_bnprep(const float* __restrict__ b, const float* __restrict__ gamma,
                         const float* __restrict__ beta, const float* __restrict__ mean,
                         const float* __restrict__ var, float* __restrict__ sa,
                         float* __restrict__ sb, int OC) {
  int i = blockIdx.x * blockDim.x + threadIdx.x;
  if (i < OC) {
    float s = gamma[i] * rsqrtf(var[i] + LN_EPS);
    sa[i] = s;
    sb[i] = (b[i] - mean[i]) * s + beta[i];
  }
}

// ---------------------------------------------------------------- encoder stage 1
// h = gelu(LN(x*W1 + b1)*g + be) written as f16.  One block (256 thr) per row, H=512.
__global__ __launch_bounds__(256)
void k_enc1(const float* __restrict__ x, const float* __restrict__ W1, const float* __restrict__ b1,
            const float* __restrict__ g, const float* __restrict__ be, h16* __restrict__ out16, int H) {
  int row = blockIdx.x;
  float xv = x[row];
  int t0 = threadIdx.x, t1 = threadIdx.x + 256;
  float v0 = xv * W1[t0] + b1[t0];
  float v1 = xv * W1[t1] + b1[t1];
  __shared__ float red[256];
  red[threadIdx.x] = v0 + v1;
  __syncthreads();
  for (int o = 128; o > 0; o >>= 1) { if (threadIdx.x < o) red[threadIdx.x] += red[threadIdx.x + o]; __syncthreads(); }
  float mu = red[0] / (float)H;
  __syncthreads();
  float d0 = v0 - mu, d1 = v1 - mu;
  red[threadIdx.x] = d0 * d0 + d1 * d1;
  __syncthreads();
  for (int o = 128; o > 0; o >>= 1) { if (threadIdx.x < o) red[threadIdx.x] += red[threadIdx.x + o]; __syncthreads(); }
  float rstd = rsqrtf(red[0] / (float)H + LN_EPS);
  float y0 = d0 * rstd * g[t0] + be[t0];
  float y1 = d1 * rstd * g[t1] + be[t1];
  out16[(size_t)row * H + t0] = (h16)gelu_exact(y0);
  out16[(size_t)row * H + t1] = (h16)gelu_exact(y1);
}

// ---------------------------------------------------------------- LayerNorm (dual output)
__global__ __launch_bounds__(256)
void k_ln(const float* __restrict__ in, const float* __restrict__ g, const float* __restrict__ be,
          float* __restrict__ out32, h16* __restrict__ out16, int H) {
  int row = blockIdx.x;
  const float* r = in + (size_t)row * H;
  int t0 = threadIdx.x, t1 = threadIdx.x + 256;
  float v0 = r[t0], v1 = r[t1];
  __shared__ float red[256];
  red[threadIdx.x] = v0 + v1;
  __syncthreads();
  for (int o = 128; o > 0; o >>= 1) { if (threadIdx.x < o) red[threadIdx.x] += red[threadIdx.x + o]; __syncthreads(); }
  float mu = red[0] / (float)H;
  __syncthreads();
  float d0 = v0 - mu, d1 = v1 - mu;
  red[threadIdx.x] = d0 * d0 + d1 * d1;
  __syncthreads();
  for (int o = 128; o > 0; o >>= 1) { if (threadIdx.x < o) red[threadIdx.x] += red[threadIdx.x + o]; __syncthreads(); }
  float rstd = rsqrtf(red[0] / (float)H + LN_EPS);
  float y0 = d0 * rstd * g[t0] + be[t0];
  float y1 = d1 * rstd * g[t1] + be[t1];
  size_t base = (size_t)row * H;
  if (out32) { out32[base + t0] = y0; out32[base + t1] = y1; }
  if (out16) { out16[base + t0] = (h16)y0; out16[base + t1] = (h16)y1; }
}

// ---------------------------------------------------------------- softmax row (N=512) -> f16
__global__ __launch_bounds__(256)
void k_softmax(const float* __restrict__ in, h16* __restrict__ out, int N) {
  int row = blockIdx.x;
  const float* r = in + (size_t)row * N;
  int t0 = threadIdx.x, t1 = threadIdx.x + 256;
  float v0 = r[t0], v1 = r[t1];
  __shared__ float red[256];
  red[threadIdx.x] = fmaxf(v0, v1);
  __syncthreads();
  for (int o = 128; o > 0; o >>= 1) { if (threadIdx.x < o) red[threadIdx.x] = fmaxf(red[threadIdx.x], red[threadIdx.x + o]); __syncthreads(); }
  float mx = red[0];
  __syncthreads();
  float e0 = __expf(v0 - mx), e1 = __expf(v1 - mx);
  red[threadIdx.x] = e0 + e1;
  __syncthreads();
  for (int o = 128; o > 0; o >>= 1) { if (threadIdx.x < o) red[threadIdx.x] += red[threadIdx.x + o]; __syncthreads(); }
  float inv = 1.0f / red[0];
  out[(size_t)row * N + t0] = (h16)(e0 * inv);
  out[(size_t)row * N + t1] = (h16)(e1 * inv);
}

// ---------------------------------------------------------------- generic WMMA GEMM
// D = ep(alpha*A@B [+C32(accum)] [*scale(n)] [+shift(n)] [+resid])
// Block tile 128x64, 8 waves each computing 32x32 (4 WMMAs/chunk, 2x A/B frag reuse).
// Double-buffered LDS staging: global loads for chunk k+1 overlap WMMAs on chunk k.
// A f16 [M,K] rows (lda); B f16: transB==0 -> B[k*ldb+n], transB==1 -> B[n*ldb+k].
// Batched: z -> (z1 = z/bz2, z2 = z%bz2); offsets z1*s?1 + z2*s?2.
// rowShift/seqLen implement conv1d shifted-A with zero padding at sequence edges.
__global__ __launch_bounds__(256)
void k_gemm(const h16* __restrict__ A, const h16* __restrict__ B,
            const float* __restrict__ scale, const float* __restrict__ shift,
            const float* __restrict__ resid,
            float* __restrict__ C32, h16* __restrict__ C16,
            int M, int N, int K, int lda, int ldb, int ldc,
            long long sA1, long long sA2, long long sB1, long long sB2,
            long long sC1, long long sC2,
            int bz2, int transB, int rowShift, int seqLen,
            int accumIn, int act, float alpha) {
  __shared__ h16 ldsA[2][128 * 32];   // [buf][m][k]
  __shared__ h16 ldsBt[2][64 * 40];   // [buf][n][k], padded stride 40 halves (80B, 16B aligned)

  int z  = blockIdx.z;
  int z1 = z / bz2, z2 = z % bz2;
  const h16* Ab = A + (size_t)(z1 * sA1 + z2 * sA2);
  const h16* Bb = B + (size_t)(z1 * sB1 + z2 * sB2);
  long long cOff = z1 * sC1 + z2 * sC2;

  int tid  = threadIdx.x;
  int lane = tid & 31;
  int w    = tid >> 5;
  int wm   = w >> 1, wn = w & 1;     // 4x2 wave grid over 128x64 tile
  int blockM = blockIdx.y * 128, blockN = blockIdx.x * 64;

  v8f acc00 = {0.f,0.f,0.f,0.f,0.f,0.f,0.f,0.f};
  v8f acc01 = acc00, acc10 = acc00, acc11 = acc00;

  // ---- hoisted per-thread staging pointers
  int am = tid >> 1;                 // 0..127 (A row within tile)
  int ak = (tid & 1) * 16;           // 0 or 16 (16 halves per thread)
  int gm = blockM + am;
  bool okA = true;
  int srcRow = gm;
  if (seqLen) {
    int s = gm % seqLen + rowShift;
    okA = (s >= 0) && (s < seqLen);
    srcRow = gm + rowShift;
  }
  const h16* Ap = Ab + (size_t)srcRow * lda + ak;        // + k0 per chunk

  int bkT0 = tid >> 3, bnT0 = (tid & 7) * 8;             // transB==0 staging
  int bnT1 = tid >> 2, bkT1 = (tid & 3) * 8;             // transB==1 staging
  const h16* Bp = transB ? (Bb + (size_t)(blockN + bnT1) * ldb + bkT1)
                         : (Bb + (size_t)bkT0 * ldb + blockN + bnT0);

  int lbase = (lane >= 16) ? 8 : 0;
  int laneN = lane & 15;
  int aRow0 = wm * 32 + laneN, aRow1 = aRow0 + 16;
  int nloc0 = wn * 32 + laneN, nloc1 = nloc0 + 16;

  // ---- preload chunk 0 into buffer 0
  {
    v8h a0 = zero8(), a1 = zero8();
    if (okA) { a0 = *(const v8h*)Ap; a1 = *(const v8h*)(Ap + 8); }
    *(v8h*)&ldsA[0][am * 32 + ak]     = a0;
    *(v8h*)&ldsA[0][am * 32 + ak + 8] = a1;
    if (!transB) {
      v8h bv = *(const v8h*)Bp;
#pragma unroll
      for (int i = 0; i < 8; ++i) ldsBt[0][(bnT0 + i) * 40 + bkT0] = bv[i];
    } else {
      *(v8h*)&ldsBt[0][bnT1 * 40 + bkT1] = *(const v8h*)Bp;
    }
  }
  __syncthreads();

  for (int k0 = 0; k0 < K; k0 += 32) {
    int cur = (k0 >> 5) & 1, nxt = cur ^ 1;
    bool more = (k0 + 32) < K;

    // ---- issue next-chunk global loads (latency hidden behind WMMAs)
    v8h a0n = zero8(), a1n = zero8(), bvn = zero8();
    if (more) {
      if (okA) {
        a0n = *(const v8h*)(Ap + k0 + 32);
        a1n = *(const v8h*)(Ap + k0 + 40);
      }
      bvn = transB ? *(const v8h*)(Bp + k0 + 32)
                   : *(const v8h*)(Bp + (size_t)(k0 + 32) * ldb);
      if (k0 + 64 < K) __builtin_prefetch(Ap + k0 + 64, 0, 3);
    }

    // ---- fragments + 4 WMMAs (2x2 reuse)
    v16h aF0, aF1, bF0, bF1;
    {
      v8h lo = *(v8h*)&ldsA[cur][aRow0 * 32 + lbase];
      v8h hi = *(v8h*)&ldsA[cur][aRow0 * 32 + 16 + lbase];
      aF0 = combine16(lo, hi);
      lo = *(v8h*)&ldsA[cur][aRow1 * 32 + lbase];
      hi = *(v8h*)&ldsA[cur][aRow1 * 32 + 16 + lbase];
      aF1 = combine16(lo, hi);
      lo = *(v8h*)&ldsBt[cur][nloc0 * 40 + lbase];
      hi = *(v8h*)&ldsBt[cur][nloc0 * 40 + 16 + lbase];
      bF0 = combine16(lo, hi);
      lo = *(v8h*)&ldsBt[cur][nloc1 * 40 + lbase];
      hi = *(v8h*)&ldsBt[cur][nloc1 * 40 + 16 + lbase];
      bF1 = combine16(lo, hi);
    }
    acc00 = __builtin_amdgcn_wmma_f32_16x16x32_f16(false, aF0, false, bF0, (short)0, acc00, false, false);
    acc01 = __builtin_amdgcn_wmma_f32_16x16x32_f16(false, aF0, false, bF1, (short)0, acc01, false, false);
    acc10 = __builtin_amdgcn_wmma_f32_16x16x32_f16(false, aF1, false, bF0, (short)0, acc10, false, false);
    acc11 = __builtin_amdgcn_wmma_f32_16x16x32_f16(false, aF1, false, bF1, (short)0, acc11, false, false);

    // ---- commit next chunk to alternate buffer
    if (more) {
      *(v8h*)&ldsA[nxt][am * 32 + ak]     = a0n;
      *(v8h*)&ldsA[nxt][am * 32 + ak + 8] = a1n;
      if (!transB) {
#pragma unroll
        for (int i = 0; i < 8; ++i) ldsBt[nxt][(bnT0 + i) * 40 + bkT0] = bvn[i];
      } else {
        *(v8h*)&ldsBt[nxt][bnT1 * 40 + bkT1] = bvn;
      }
      __syncthreads();
    }
  }

  // ---- epilogue
#pragma unroll
  for (int sm = 0; sm < 2; ++sm) {
#pragma unroll
    for (int sn = 0; sn < 2; ++sn) {
      v8f acc = sm ? (sn ? acc11 : acc10) : (sn ? acc01 : acc00);
      int n = blockN + wn * 32 + sn * 16 + laneN;
      if (n >= N) continue;
#pragma unroll
      for (int i = 0; i < 8; ++i) {
        int m = blockM + wm * 32 + sm * 16 + i + lbase;
        if (m >= M) continue;
        size_t idx = (size_t)(cOff + (long long)m * ldc + n);
        float v = acc[i] * alpha;
        if (accumIn) v += C32[idx];
        if (scale) v *= scale[n];
        if (shift) v += shift[n];
        if (resid) v += resid[idx];
        if (act == 1) v = gelu_exact(v);
        if (C32) C32[idx] = v;
        if (C16) C16[idx] = (h16)v;
      }
    }
  }
}

// ---------------------------------------------------------------- liquid-ODE scan
// One block per direction.  h (32x512) in LDS (f16 for WMMA A); recurrent state and
// mean-pool sum live in per-lane f32 registers.  WrecT is [n][k] f16.
__global__ __launch_bounds__(256)
void k_scan(const float* __restrict__ preF, const float* __restrict__ preB,
            const h16* __restrict__ WrecTf, const h16* __restrict__ WrecTb,
            const float* __restrict__ tauF, const float* __restrict__ tauB,
            float* __restrict__ pooled, int S, int H, float dt) {
  int dir = blockIdx.x;
  const float* pre = dir ? preB : preF;
  const h16* WrecT = dir ? WrecTb : WrecTf;
  const float* tau = dir ? tauB : tauF;

  __shared__ h16 hlds[32 * 512];   // [batch][hidden] f16 = 32KB

  int tid = threadIdx.x, lane = tid & 31, w = tid >> 5;
  {
    v8h zz = zero8();
    for (int i = tid; i < 32 * 512 / 8; i += 256) ((v8h*)hlds)[i] = zz;
  }
  int tileM = w >> 2;                 // 0..1 (batch rows 0-15 / 16-31)
  int nTileBase = (w & 3) * 8;        // 8 n-tiles per wave
  int lbase = (lane >= 16) ? 8 : 0;
  int mRowFrag = tileM * 16 + (lane & 15);

  float hreg[8][8];
  float sreg[8][8];
#pragma unroll
  for (int i = 0; i < 8; ++i)
#pragma unroll
    for (int e = 0; e < 8; ++e) { hreg[i][e] = 0.f; sreg[i][e] = 0.f; }
  __syncthreads();

  for (int t = 0; t < S; ++t) {
    int ts = dir ? (S - 1 - t) : t;
#pragma unroll
    for (int i = 0; i < 8; ++i) {
      int nBase = (nTileBase + i) * 16;
      v8f acc = {0.f, 0.f, 0.f, 0.f, 0.f, 0.f, 0.f, 0.f};
#pragma unroll 4
      for (int kc = 0; kc < 512; kc += 32) {
        v8h alo = *(v8h*)&hlds[mRowFrag * 512 + kc + lbase];
        v8h ahi = *(v8h*)&hlds[mRowFrag * 512 + kc + 16 + lbase];
        v16h a = combine16(alo, ahi);
        const h16* bp = WrecT + (size_t)(nBase + (lane & 15)) * H + kc;
        v8h blo = *(const v8h*)(bp + lbase);
        v8h bhi = *(const v8h*)(bp + 16 + lbase);
        v16h b = combine16(blo, bhi);
        acc = __builtin_amdgcn_wmma_f32_16x16x32_f16(false, a, false, b, (short)0, acc, false, false);
      }
      int n = nBase + (lane & 15);
      float tauv = tau[n];
#pragma unroll
      for (int e = 0; e < 8; ++e) {
        int m = tileM * 16 + e + lbase;   // batch index
        float p = pre[((size_t)m * S + ts) * H + n];
        float f = tanhf(p + acc[e]);
        float hp = hreg[i][e];
        float hn = hp + dt * (-hp / tauv + f);
        hn = fminf(10.0f, fmaxf(-10.0f, hn));
        hreg[i][e] = hn;
        sreg[i][e] += hn;
      }
    }
    __syncthreads();
#pragma unroll
    for (int i = 0; i < 8; ++i) {
      int n = (nTileBase + i) * 16 + (lane & 15);
#pragma unroll
      for (int e = 0; e < 8; ++e) {
        int m = tileM * 16 + e + lbase;
        hlds[m * 512 + n] = (h16)hreg[i][e];
      }
    }
    __syncthreads();
  }
  float invS = 1.0f / (float)S;
#pragma unroll
  for (int i = 0; i < 8; ++i) {
    int n = (nTileBase + i) * 16 + (lane & 15);
#pragma unroll
    for (int e = 0; e < 8; ++e) {
      int m = tileM * 16 + e + lbase;
      pooled[(size_t)m * (2 * H) + dir * H + n] = sreg[i][e] * invS;
    }
  }
}

// ---------------------------------------------------------------- final FC (2H -> 1)
__global__ __launch_bounds__(256)
void k_fc(const float* __restrict__ pooled, const float* __restrict__ W,
          const float* __restrict__ b, float* __restrict__ out, int D) {
  int row = blockIdx.x;
  float s = 0.f;
  for (int j = threadIdx.x; j < D; j += blockDim.x) s += pooled[(size_t)row * D + j] * W[j];
  __shared__ float red[256];
  red[threadIdx.x] = s;
  __syncthreads();
  for (int o = 128; o > 0; o >>= 1) { if (threadIdx.x < o) red[threadIdx.x] += red[threadIdx.x + o]; __syncthreads(); }
  if (threadIdx.x == 0) out[row] = red[0] + b[0];
}

// ================================================================ host orchestration
extern "C" void kernel_launch(void* const* d_in, const int* in_sizes, int n_in,
                              void* d_out, int out_size, void* d_ws, size_t ws_size,
                              hipStream_t stream) {
  (void)in_sizes; (void)n_in; (void)out_size; (void)ws_size;
  const int Bb = 32, S = 512, H = 512, NH = 4, DH = 128, F = 2048;
  const int R = Bb * S;  // 16384 rows
  const float DT = 0.1f;

  auto fin = [&](int i) { return (const float*)d_in[i]; };
  // pytree (sorted-key) flatten order: cnn[0..2]{W,b,beta,gamma,mean,var}, enc{W1,W2,b1,b2,be1,be2,g1,g2},
  // fc{W,b}, lq_bwd{Win,Wrec,bin,tau}, lq_fwd{...}, tx[0..1]{Wf1,Wf2,Wk,Wo,Wq,Wv,be1,be2,bf1,bf2,bk,bo,bq,bv,g1,g2}, x
  const float *cnnW[3], *cnnB[3], *cnnBeta[3], *cnnGamma[3], *cnnMean[3], *cnnVar[3];
  for (int l = 0; l < 3; ++l) {
    int b0 = l * 6;
    cnnW[l] = fin(b0 + 0); cnnB[l] = fin(b0 + 1); cnnBeta[l] = fin(b0 + 2);
    cnnGamma[l] = fin(b0 + 3); cnnMean[l] = fin(b0 + 4); cnnVar[l] = fin(b0 + 5);
  }
  const float* encW1 = fin(18); const float* encW2 = fin(19);
  const float* encB1 = fin(20); const float* encB2 = fin(21);
  const float* encBe1 = fin(22); const float* encBe2 = fin(23);
  const float* encG1 = fin(24); const float* encG2 = fin(25);
  const float* fcW = fin(26); const float* fcB = fin(27);
  const float* lqbWin = fin(28); const float* lqbWrec = fin(29);
  const float* lqbBin = fin(30); const float* lqbTau = fin(31);
  const float* lqfWin = fin(32); const float* lqfWrec = fin(33);
  const float* lqfBin = fin(34); const float* lqfTau = fin(35);
  const float *txWf1[2], *txWf2[2], *txWk[2], *txWo[2], *txWq[2], *txWv[2];
  const float *txBe1[2], *txBe2[2], *txBf1[2], *txBf2[2], *txBk[2], *txBo[2], *txBq[2], *txBv[2], *txG1[2], *txG2[2];
  for (int l = 0; l < 2; ++l) {
    int b0 = 36 + l * 16;
    txWf1[l] = fin(b0 + 0); txWf2[l] = fin(b0 + 1); txWk[l] = fin(b0 + 2); txWo[l] = fin(b0 + 3);
    txWq[l] = fin(b0 + 4); txWv[l] = fin(b0 + 5); txBe1[l] = fin(b0 + 6); txBe2[l] = fin(b0 + 7);
    txBf1[l] = fin(b0 + 8); txBf2[l] = fin(b0 + 9); txBk[l] = fin(b0 + 10); txBo[l] = fin(b0 + 11);
    txBq[l] = fin(b0 + 12); txBv[l] = fin(b0 + 13); txG1[l] = fin(b0 + 14); txG2[l] = fin(b0 + 15);
  }
  const float* x = fin(68);

  // ---------------- workspace layout
  char* ws = (char*)d_ws;
  size_t off = 0;
  auto alloc = [&](size_t bytes) -> char* {
    char* p = ws + off;
    off += (bytes + 255) & ~(size_t)255;
    return p;
  };
  h16* w2_16 = (h16*)alloc((size_t)H * H * 2);
  int icArr[3] = {H, 128, 256}, ocArr[3] = {128, 256, H};
  h16* wt[3]; float* saP[3]; float* sbP[3];
  for (int l = 0; l < 3; ++l) {
    wt[l]  = (h16*)alloc((size_t)3 * icArr[l] * ocArr[l] * 2);
    saP[l] = (float*)alloc((size_t)ocArr[l] * 4);
    sbP[l] = (float*)alloc((size_t)ocArr[l] * 4);
  }
  h16 *wq16[2], *wk16[2], *wv16[2], *wo16[2], *wf1_16[2], *wf2_16[2];
  for (int l = 0; l < 2; ++l) {
    wq16[l] = (h16*)alloc((size_t)H * H * 2);  wk16[l] = (h16*)alloc((size_t)H * H * 2);
    wv16[l] = (h16*)alloc((size_t)H * H * 2);  wo16[l] = (h16*)alloc((size_t)H * H * 2);
    wf1_16[l] = (h16*)alloc((size_t)H * F * 2); wf2_16[l] = (h16*)alloc((size_t)F * H * 2);
  }
  h16* winF16 = (h16*)alloc((size_t)H * H * 2);
  h16* winB16 = (h16*)alloc((size_t)H * H * 2);
  h16* wrtF16 = (h16*)alloc((size_t)H * H * 2);
  h16* wrtB16 = (h16*)alloc((size_t)H * H * 2);

  float* bufX = (float*)alloc((size_t)R * H * 4);
  float* bufY = (float*)alloc((size_t)R * H * 4);
  h16* x16a = (h16*)alloc((size_t)R * H * 2);
  h16* x16b = (h16*)alloc((size_t)R * H * 2);
  h16* q16  = (h16*)alloc((size_t)R * H * 2);
  h16* k16  = (h16*)alloc((size_t)R * H * 2);
  h16* v16  = (h16*)alloc((size_t)R * H * 2);
  h16* o16  = (h16*)alloc((size_t)R * H * 2);
  h16* ffn16 = (h16*)alloc((size_t)R * F * 2);
  float* scores = (float*)alloc((size_t)Bb * NH * S * S * 4);
  h16* attn16 = (h16*)alloc((size_t)Bb * NH * S * S * 2);
  float* preF = (float*)alloc((size_t)R * H * 4);
  float* preB = (float*)alloc((size_t)R * H * 4);
  float* pooled = (float*)alloc((size_t)Bb * 2 * H * 4);

  auto cvt = [&](const float* src, h16* dst, size_t n) {
    k_f32_to_f16<<<dim3((unsigned)((n + 255) / 256)), dim3(256), 0, stream>>>(src, dst, (int)n);
  };
  auto gemm = [&](const h16* A, const h16* B, const float* scale, const float* shift,
                  const float* resid, float* C32, h16* C16,
                  int M, int N, int K, int lda, int ldb, int ldc,
                  long long sA1, long long sA2, long long sB1, long long sB2,
                  long long sC1, long long sC2, int nz, int bz2,
                  int transB, int rowShift, int seqLen, int accumIn, int act, float alpha) {
    dim3 g(N / 64, M / 128, nz), blk(256, 1, 1);
    k_gemm<<<g, blk, 0, stream>>>(A, B, scale, shift, resid, C32, C16, M, N, K,
                                  lda, ldb, ldc, sA1, sA2, sB1, sB2, sC1, sC2,
                                  bz2, transB, rowShift, seqLen, accumIn, act, alpha);
  };

  // ---------------- weight prep (every call; deterministic)
  cvt(encW2, w2_16, (size_t)H * H);
  for (int l = 0; l < 3; ++l) {
    int total = ocArr[l] * icArr[l] * 3;
    k_convprep<<<dim3((total + 255) / 256), dim3(256), 0, stream>>>(cnnW[l], wt[l], ocArr[l], icArr[l]);
    k_bnprep<<<dim3((ocArr[l] + 255) / 256), dim3(256), 0, stream>>>(
        cnnB[l], cnnGamma[l], cnnBeta[l], cnnMean[l], cnnVar[l], saP[l], sbP[l], ocArr[l]);
  }
  for (int l = 0; l < 2; ++l) {
    cvt(txWq[l], wq16[l], (size_t)H * H);  cvt(txWk[l], wk16[l], (size_t)H * H);
    cvt(txWv[l], wv16[l], (size_t)H * H);  cvt(txWo[l], wo16[l], (size_t)H * H);
    cvt(txWf1[l], wf1_16[l], (size_t)H * F); cvt(txWf2[l], wf2_16[l], (size_t)F * H);
  }
  cvt(lqfWin, winF16, (size_t)H * H);
  cvt(lqbWin, winB16, (size_t)H * H);
  k_trans_f32_to_f16<<<dim3((H * H + 255) / 256), dim3(256), 0, stream>>>(lqfWrec, wrtF16, H, H);
  k_trans_f32_to_f16<<<dim3((H * H + 255) / 256), dim3(256), 0, stream>>>(lqbWrec, wrtB16, H, H);

  // ---------------- 1. encoder
  k_enc1<<<R, 256, 0, stream>>>(x, encW1, encB1, encG1, encBe1, x16a, H);
  gemm(x16a, w2_16, nullptr, encB2, nullptr, bufX, nullptr,
       R, H, H, H, H, H, 0, 0, 0, 0, 0, 0, 1, 1, 0, 0, 0, 0, 0, 1.0f);
  k_ln<<<R, 256, 0, stream>>>(bufX, encG2, encBe2, nullptr, x16a, H);

  // ---------------- 2. graduated CNN (3 shifted GEMMs per layer, BN+GELU epilogue)
  {
    const h16* cin = x16a;
    h16* cout16[3] = {x16b, x16a, x16b};
    for (int l = 0; l < 3; ++l) {
      int IC = icArr[l], OC = ocArr[l];
      for (int t = 0; t < 3; ++t) {
        bool last = (t == 2);
        gemm(cin, wt[l] + (size_t)t * IC * OC,
             last ? saP[l] : nullptr, last ? sbP[l] : nullptr, nullptr,
             bufX, last ? cout16[l] : nullptr,
             R, OC, IC, IC, OC, OC, 0, 0, 0, 0, 0, 0, 1, 1,
             0, t - 1, S, (t > 0) ? 1 : 0, last ? 1 : 0, 1.0f);
      }
      cin = cout16[l];
    }
  }
  // now: bufX = h (f32), x16b = h (f16)
  float* h32 = bufX;
  float* scr = bufY;
  h16* h16c = x16b;

  // ---------------- 3. transformer x2
  const long long sBH = (long long)S * H, sSS = (long long)S * S;
  for (int l = 0; l < 2; ++l) {
    gemm(h16c, wq16[l], nullptr, txBq[l], nullptr, nullptr, q16,
         R, H, H, H, H, H, 0, 0, 0, 0, 0, 0, 1, 1, 0, 0, 0, 0, 0, 1.0f);
    gemm(h16c, wk16[l], nullptr, txBk[l], nullptr, nullptr, k16,
         R, H, H, H, H, H, 0, 0, 0, 0, 0, 0, 1, 1, 0, 0, 0, 0, 0, 1.0f);
    gemm(h16c, wv16[l], nullptr, txBv[l], nullptr, nullptr, v16,
         R, H, H, H, H, H, 0, 0, 0, 0, 0, 0, 1, 1, 0, 0, 0, 0, 0, 1.0f);
    // scores = Q @ K^T / sqrt(DH), batched over (b, head)
    gemm(q16, k16, nullptr, nullptr, nullptr, scores, nullptr,
         S, S, DH, H, H, S, sBH, DH, sBH, DH, (long long)NH * sSS, sSS,
         Bb * NH, NH, 1, 0, 0, 0, 0, 0.08838834764831845f);
    k_softmax<<<Bb * NH * S, 256, 0, stream>>>(scores, attn16, S);
    // o = attn @ V
    gemm(attn16, v16, nullptr, nullptr, nullptr, nullptr, o16,
         S, DH, S, S, H, H, (long long)NH * sSS, sSS, sBH, DH, sBH, DH,
         Bb * NH, NH, 0, 0, 0, 0, 0, 1.0f);
    // h = LN(h + o@Wo + bo)
    gemm(o16, wo16[l], nullptr, txBo[l], h32, scr, nullptr,
         R, H, H, H, H, H, 0, 0, 0, 0, 0, 0, 1, 1, 0, 0, 0, 0, 0, 1.0f);
    k_ln<<<R, 256, 0, stream>>>(scr, txG1[l], txBe1[l], h32, h16c, H);
    // FFN
    gemm(h16c, wf1_16[l], nullptr, txBf1[l], nullptr, nullptr, ffn16,
         R, F, H, H, F, F, 0, 0, 0, 0, 0, 0, 1, 1, 0, 0, 0, 0, 1, 1.0f);
    gemm(ffn16, wf2_16[l], nullptr, txBf2[l], h32, scr, nullptr,
         R, H, F, F, H, H, 0, 0, 0, 0, 0, 0, 1, 1, 0, 0, 0, 0, 0, 1.0f);
    k_ln<<<R, 256, 0, stream>>>(scr, txG2[l], txBe2[l], h32, h16c, H);
  }

  // ---------------- 4. liquid scans: pre = h@Win + bin, then persistent WMMA recurrence
  gemm(h16c, winF16, nullptr, lqfBin, nullptr, preF, nullptr,
       R, H, H, H, H, H, 0, 0, 0, 0, 0, 0, 1, 1, 0, 0, 0, 0, 0, 1.0f);
  gemm(h16c, winB16, nullptr, lqbBin, nullptr, preB, nullptr,
       R, H, H, H, H, H, 0, 0, 0, 0, 0, 0, 1, 1, 0, 0, 0, 0, 0, 1.0f);
  k_scan<<<2, 256, 0, stream>>>(preF, preB, wrtF16, wrtB16, lqfTau, lqbTau, pooled, S, H, DT);

  // ---------------- 5. FC head
  k_fc<<<Bb, 256, 0, stream>>>(pooled, fcW, fcB, (float*)d_out, 2 * H);
}
